// QuantLinearWithStats_68616397521439
// MI455X (gfx1250) — 8-bit matrix operand order repaired and checked by reading, compiled, not run on hardware
//
#include <hip/hip_runtime.h>


typedef __attribute__((ext_vector_type(8))) int v8i;
typedef int v4i_vec __attribute__((__vector_size__(16)));

#define M_DIM 4096
#define N_DIM 4096
#define K_DIM 4096

#define TM 128
#define TN 256
#define TK 64
#define LDS_STRIDE 80

#define ASG __attribute__((address_space(1)))
#define ASL __attribute__((address_space(3)))

#if defined(__gfx1250__) && __has_builtin(__builtin_amdgcn_global_load_async_to_lds_b128)
#define USE_ASYNC 1
#else
#define USE_ASYNC 0
#endif

#if USE_ASYNC
#define COPY16(gsrc, ldst)                                                    \
    __builtin_amdgcn_global_load_async_to_lds_b128(                           \
        (ASG v4i_vec*)(gsrc), (ASL v4i_vec*)(ldst), 0, 0)
#if __has_builtin(__builtin_amdgcn_s_wait_asynccnt)
#define WAIT_ASYNC() __builtin_amdgcn_s_wait_asynccnt(0)
#else
#define WAIT_ASYNC() asm volatile("s_wait_asynccnt 0x0" ::: "memory")
#endif
#else
#define COPY16(gsrc, ldst) *(int4*)(ldst) = *(const int4*)(gsrc)
#define WAIT_ASYNC()
#endif

__global__ __launch_bounds__(256) void quant_ternary_kernel(
    const float* __restrict__ in, signed char* __restrict__ out,
    float th, int n_chunks16)
{
    int idx = blockIdx.x * blockDim.x + threadIdx.x;
    if (idx >= n_chunks16) return;

    const float4* src = (const float4*)in + (size_t)idx * 4;
    signed char q[16];
#pragma unroll
    for (int i = 0; i < 4; ++i) {
        float4 v = src[i];
        float f[4] = {v.x, v.y, v.z, v.w};
#pragma unroll
        for (int j = 0; j < 4; ++j) {
            float x = f[j];
            q[i * 4 + j] = (signed char)(x > th ? 1 : (x < -th ? -1 : 0));
        }
    }
    *(int4*)(out + (size_t)idx * 16) = *(const int4*)q;
}

__global__ __launch_bounds__(256) void ternary_gemm_wmma_kernel(
    const signed char* __restrict__ Aq,
    const signed char* __restrict__ Bq,
    float* __restrict__ Out,
    float scale)
{
    __shared__ signed char As[2][TM * LDS_STRIDE];
    __shared__ signed char Bs[2][TN * LDS_STRIDE];

    const int tid  = threadIdx.x;
    const int lane = tid & 31;
    const int wave = tid >> 5;
    const int wm   = (wave & 1) * 64;
    const int wn   = (wave >> 1) * 64;

    const int tileM = blockIdx.y * TM;
    const int tileN = blockIdx.x * TN;

    const int half    = lane >> 4;
    const int sublane = lane & 15;

    v8i acc[4][4] = {};

    auto copy_tile = [&](int b, int k0) {
#pragma unroll
        for (int j = 0; j < 2; ++j) {
            int c = j * 256 + tid;
            int row = c >> 2, col = (c & 3) * 16;
            COPY16(Aq + (size_t)(tileM + row) * K_DIM + k0 + col,
                   &As[b][row * LDS_STRIDE + col]);
        }
#pragma unroll
        for (int j = 0; j < 4; ++j) {
            int c = j * 256 + tid;
            int row = c >> 2, col = (c & 3) * 16;
            COPY16(Bq + (size_t)(tileN + row) * K_DIM + k0 + col,
                   &Bs[b][row * LDS_STRIDE + col]);
        }
    };

    copy_tile(0, 0);
    WAIT_ASYNC();
    __syncthreads();

    int buf = 0;
    for (int k0 = 0; k0 < K_DIM; k0 += TK) {
        if (k0 + TK < K_DIM) copy_tile(buf ^ 1, k0 + TK);

        v8i afrag[4];
        const int akb = half * 16;
#pragma unroll
        for (int f = 0; f < 4; ++f) {
            const signed char* p =
                &As[buf][(wm + f * 16 + sublane) * LDS_STRIDE + akb];
            union { v8i v; long long q[4]; } u;
            u.q[0] = *(const long long*)(p);
            u.q[1] = *(const long long*)(p + 8);
            u.q[2] = *(const long long*)(p + 32);
            u.q[3] = *(const long long*)(p + 40);
            afrag[f] = u.v;
        }

        v8i bfrag[4];
        const int bkb = half * 16;
#pragma unroll
        for (int g = 0; g < 4; ++g) {
            const signed char* p =
                &Bs[buf][(wn + g * 16 + sublane) * LDS_STRIDE + bkb];
            union { v8i v; int4 q[2]; } u;
            u.q[0] = *(const int4*)(p);
            u.q[1] = *(const int4*)(p + 32);
            bfrag[g] = u.v;
        }

#pragma unroll
        for (int f = 0; f < 4; ++f)
#pragma unroll
            for (int g = 0; g < 4; ++g)
                acc[f][g] = __builtin_amdgcn_wmma_i32_16x16x64_iu8(
                     true, afrag[f],
                     true, bfrag[g],
                    acc[f][g],
                     false,  false);

        WAIT_ASYNC();
        __syncthreads();
        buf ^= 1;
    }

#pragma unroll
    for (int f = 0; f < 4; ++f) {
        int mbase = tileM + wm + f * 16 + half * 8;
#pragma unroll
        for (int g = 0; g < 4; ++g) {
            int n = tileN + wn + g * 16 + sublane;
#pragma unroll
            for (int v = 0; v < 8; ++v) {
                Out[(size_t)(mbase + v) * N_DIM + n] = scale * (float)acc[f][g][v];
            }
        }
    }
}

extern "C" void kernel_launch(void* const* d_in, const int* in_sizes, int n_in,
                              void* d_out, int out_size, void* d_ws, size_t ws_size,
                              hipStream_t stream) {
    const float* x = (const float*)d_in[0];
    const float* w = (const float*)d_in[1];
    float* out = (float*)d_out;

    signed char* xq = (signed char*)d_ws;
    signed char* wq = xq + (size_t)M_DIM * K_DIM;

    const int n16 = (M_DIM * K_DIM) / 16;
    quant_ternary_kernel<<<n16 / 256, 256, 0, stream>>>(x, xq, 0.33f, n16);
    quant_ternary_kernel<<<n16 / 256, 256, 0, stream>>>(w, wq, 0.20f, n16);

    dim3 grid(N_DIM / TN, M_DIM / TM);
    ternary_gemm_wmma_kernel<<<grid, 256, 0, stream>>>(
        xq, wq, out, 0.5f * 0.6f  );
}
